// AttnReadout_52055003627521
// MI455X (gfx1250) — matrix operand order repaired and checked by reading, not compiled, not run on hardware
//
#include <hip/hip_runtime.h>

// ---------------------------------------------------------------------------
// AttnReadout pipeline for MI455X (gfx1250, wave32, WMMA + async-to-LDS)
//   N=204800 nodes, D=256, H=256, O=512, B=4096 segments
// Main GEMMs: v_wmma_f32_16x16x32_bf16, f32 accumulate.
// BN folded into packed weights: x@W = feat@(scale*W) + shift@W, so the A
// operand is raw bf16 feat -> staged with GLOBAL_LOAD_ASYNC_TO_LDS_B128 and
// double-buffered (s_wait_asynccnt) to overlap copy with WMMA.
// ---------------------------------------------------------------------------

typedef __attribute__((ext_vector_type(16))) __bf16 v16bf;
typedef __attribute__((ext_vector_type(8)))  __bf16 v8bf;
typedef __attribute__((ext_vector_type(4)))  __bf16 v4bf;
typedef __attribute__((ext_vector_type(8)))  float  v8f;

#define N_NODES 204800
#define DIM     256
#define HID     256
#define OUTD    512
#define NSEG    4096
#define BN_EPS  1e-5f
#define LDA     264   // LDS A row stride in bf16 (528B = 33*16B: b128-aligned, conflict-free)
#define MT      4     // row tiles per macro-tile (64 rows)

union AB16 { v16bf v; v8bf h[2]; };

// ----------------------- async copy helpers (CDNA5) ------------------------
__device__ __forceinline__ void async_copy16(unsigned lds_off, const void* gptr) {
  asm volatile("global_load_async_to_lds_b128 %0, %1, off"
               :: "v"(lds_off), "v"((unsigned long long)(uintptr_t)gptr)
               : "memory");
}
__device__ __forceinline__ void wait_async0() {
  asm volatile("s_wait_asynccnt 0x0" ::: "memory");
}
__device__ __forceinline__ void wait_async8() {   // allow 8 outstanding (next tile)
  asm volatile("s_wait_asynccnt 0x8" ::: "memory");
}

// ------------------------- BN statistics + bf16 copy -----------------------
__global__ void k_init_sums(float* sums) {              // 1 block x 512
  sums[threadIdx.x] = 0.0f;
}

// 400 blocks x 256 thr; 512 rows/block. Thread: 4 cols (t&63)*4, rows (t>>6)+4k.
// Also writes featb = bf16(feat) (fused: feat is already streaming through).
__global__ void k_bn_stats(const float* __restrict__ feat, float* __restrict__ sum,
                           float* __restrict__ sumsq, __bf16* __restrict__ featb) {
  __shared__ float4 ps[256], pq[256];
  const int t = threadIdx.x;
  const int c4 = (t & 63) * 4;
  const int rsub = t >> 6;
  const size_t r0 = (size_t)blockIdx.x * 512 + rsub;
  float4 s = {0, 0, 0, 0}, q = {0, 0, 0, 0};
  for (int k = 0; k < 128; ++k) {
    const size_t row = r0 + (size_t)k * 4;
    const float4 f = *(const float4*)&feat[row * DIM + c4];
    s.x += f.x; s.y += f.y; s.z += f.z; s.w += f.w;
    q.x += f.x * f.x; q.y += f.y * f.y; q.z += f.z * f.z; q.w += f.w * f.w;
    v4bf bv; bv[0] = (__bf16)f.x; bv[1] = (__bf16)f.y; bv[2] = (__bf16)f.z; bv[3] = (__bf16)f.w;
    *(v4bf*)&featb[row * DIM + c4] = bv;
  }
  ps[t] = s; pq[t] = q;
  __syncthreads();
  if (t < 64) {
    float4 a = ps[t], b = ps[64 + t], c = ps[128 + t], d = ps[192 + t];
    float4 e = pq[t], f = pq[64 + t], g = pq[128 + t], h = pq[192 + t];
    const int c0 = t * 4;
    unsafeAtomicAdd(&sum[c0 + 0], a.x + b.x + c.x + d.x);
    unsafeAtomicAdd(&sum[c0 + 1], a.y + b.y + c.y + d.y);
    unsafeAtomicAdd(&sum[c0 + 2], a.z + b.z + c.z + d.z);
    unsafeAtomicAdd(&sum[c0 + 3], a.w + b.w + c.w + d.w);
    unsafeAtomicAdd(&sumsq[c0 + 0], e.x + f.x + g.x + h.x);
    unsafeAtomicAdd(&sumsq[c0 + 1], e.y + f.y + g.y + h.y);
    unsafeAtomicAdd(&sumsq[c0 + 2], e.z + f.z + g.z + h.z);
    unsafeAtomicAdd(&sumsq[c0 + 3], e.w + f.w + g.w + h.w);
  }
}

__global__ void k_bn_finalize(const float* __restrict__ sum, const float* __restrict__ sumsq,
                              const float* __restrict__ gamma, const float* __restrict__ beta,
                              float* __restrict__ scale, float* __restrict__ shift) {
  const int t = threadIdx.x;  // 1 block x 256
  const float invn = 1.0f / (float)N_NODES;
  float mean = sum[t] * invn;
  float var  = sumsq[t] * invn - mean * mean;
  float rstd = rsqrtf(var + BN_EPS);
  float sc   = gamma[t] * rstd;
  scale[t] = sc;
  shift[t] = beta[t] - mean * sc;
}

// ---- Pack f32 weight (K=256 x Nn), optionally row-scaled, into WMMA-B tiles
__global__ void k_pack_b(const float* __restrict__ W, const float* __restrict__ scale,
                         __bf16* __restrict__ out, int Nn) {
  int t = blockIdx.x * blockDim.x + threadIdx.x;
  int p = t >> 9, r = t & 511;
  int lane = r >> 4, j = r & 15;
  int n_tile = p >> 3, ks = p & 7;
  int n = n_tile * 16 + (lane & 15);
  int k = ks * 32 + (lane >> 4) * 16 + j;
  float sc = scale ? scale[k] : 1.0f;
  out[t] = (__bf16)(W[(size_t)k * Nn + n] * sc);
}

// colconst[n] = sum_k shift[k] * W[k][n]
__global__ void k_colconst(const float* __restrict__ W, const float* __restrict__ shift,
                           float* __restrict__ out, int Nn) {
  int n = blockIdx.x * blockDim.x + threadIdx.x;
  if (n >= Nn) return;
  float s = 0.f;
  for (int k = 0; k < DIM; ++k) s += shift[k] * W[(size_t)k * Nn + n];
  out[n] = s;
}

// ---------------- Segment boundaries (segment_ids sorted) ------------------
__global__ void k_seg_bounds(const int* __restrict__ seg, int* __restrict__ seg_start) {
  int i = blockIdx.x * blockDim.x + threadIdx.x;
  if (i >= N_NODES) return;
  int cur  = seg[i];
  int prev = (i == 0) ? -1 : seg[i - 1];
  for (int b = prev + 1; b <= cur; ++b) seg_start[b] = i;
  if (i == N_NODES - 1)
    for (int b = cur + 1; b <= NSEG; ++b) seg_start[b] = N_NODES;
}

// One 16x16 f32 tile, K=256 = 8 x (16x16x32 bf16 WMMA).
__device__ __forceinline__ v8f wmma_k256(const __bf16* lds_a, const __bf16* bpk, int lane) {
  v8f c = {};
  const int m  = lane & 15;
  const int hf = lane >> 4;
  #pragma unroll
  for (int ks = 0; ks < 8; ++ks) {
    AB16 a, b;
    const __bf16* arow = lds_a + m * LDA + ks * 32 + hf * 16;
    a.h[0] = *(const v8bf*)(arow);
    a.h[1] = *(const v8bf*)(arow + 8);
    const __bf16* bl = bpk + ks * 512 + lane * 16;
    b.h[0] = *(const v8bf*)(bl);
    b.h[1] = *(const v8bf*)(bl + 8);
    c = __builtin_amdgcn_wmma_f32_16x16x32_bf16(false, a.v, false, b.v, (short)0, c,
                                                false, false);
  }
  return c;
}

// ------- feat_v = feat[last_nodes] @ (scale*w_v) + (b_v + shift@w_v) -------
__global__ void __launch_bounds__(256)
k_gemm_v(const __bf16* __restrict__ featb, const int* __restrict__ last_nodes,
         const __bf16* __restrict__ wv_pk, const float* __restrict__ b_v,
         const float* __restrict__ vconst, float* __restrict__ featv) {
  __shared__ __bf16 lds_a[16 * LDA];
  __shared__ int rows[16];
  const int rt = blockIdx.x, t = threadIdx.x;   // 256 blocks x 256 thr
  if (t < 16) rows[t] = last_nodes[rt * 16 + t];
  __syncthreads();
  {
    const unsigned lbase = (unsigned)(uintptr_t)&lds_a[0];
    #pragma unroll
    for (int i = 0; i < 2; ++i) {               // 16 rows x 512B, 16B chunks
      int id = i * 256 + t;
      int row = id >> 5, c = (id & 31) * 16;
      async_copy16(lbase + row * (LDA * 2) + c,
                   (const char*)(featb + (size_t)rows[row] * DIM) + c);
    }
  }
  wait_async0();
  __syncthreads();
  const int wave = t >> 5, lane = t & 31, hf = lane >> 4;
  #pragma unroll
  for (int i = 0; i < 2; ++i) {
    int nt = wave * 2 + i;
    v8f c = wmma_k256(lds_a, wv_pk + (size_t)nt * 8 * 512, lane);
    int n = nt * 16 + (lane & 15);
    float bb = b_v[n] + vconst[n];
    #pragma unroll
    for (int j = 0; j < 8; ++j) {
      int rl = j + hf * 8;
      featv[(size_t)(rt * 16 + rl) * HID + n] = c[j] + bb;
    }
  }
}

// ---- Big GEMM compute for one 64-row macro-tile (MT=4 register-blocked) ---
__device__ __forceinline__ void gemm_u_compute(const __bf16* lds_buf, const int* sseg,
                                               float* e_part, const __bf16* __restrict__ wu_pk,
                                               const float* __restrict__ econst,
                                               const float* __restrict__ featv,
                                               const float* __restrict__ w_e, int t) {
  const int wave = t >> 5, lane = t & 31, hf = lane >> 4, m = lane & 15;
  #pragma unroll
  for (int i = 0; i < 2; ++i) {
    const int nt = wave * 2 + i;
    const __bf16* bpk = wu_pk + (size_t)nt * 8 * 512;
    v8f cacc[MT];
    #pragma unroll
    for (int mt = 0; mt < MT; ++mt) cacc[mt] = (v8f){};
    #pragma unroll
    for (int ks = 0; ks < 8; ++ks) {
      AB16 b;                                   // B fragment reused MT times
      const __bf16* bl = bpk + ks * 512 + lane * 16;
      b.h[0] = *(const v8bf*)(bl);
      b.h[1] = *(const v8bf*)(bl + 8);
      #pragma unroll
      for (int mt = 0; mt < MT; ++mt) {
        AB16 a;
        const __bf16* arow = lds_buf + (mt * 16 + m) * LDA + ks * 32 + hf * 16;
        a.h[0] = *(const v8bf*)(arow);
        a.h[1] = *(const v8bf*)(arow + 8);
        cacc[mt] = __builtin_amdgcn_wmma_f32_16x16x32_bf16(false, a.v, false, b.v,
                                                           (short)0, cacc[mt], false, false);
      }
    }
    const int n = nt * 16 + m;
    const float we = w_e[n];
    const float ec = econst[n];
    #pragma unroll
    for (int mt = 0; mt < MT; ++mt) {
      #pragma unroll
      for (int j = 0; j < 8; ++j) {
        int rl = mt * 16 + j + hf * 8;
        float u = cacc[mt][j] + featv[(size_t)sseg[rl] * HID + n] + ec;
        atomicAdd(&e_part[rl], we / (1.0f + __expf(-u)));   // ds_add_f32
      }
    }
  }
}

// e1[row] = sum_n sigmoid((feat@scaled_wu)[row,n] + econst[n] + featv[seg,n]) * w_e[n]
// 1600 blocks x 256 thr; 128 rows/block = 2 async-double-buffered macro-tiles.
__global__ void __launch_bounds__(256)
k_gemm_u_e(const __bf16* __restrict__ featb, const __bf16* __restrict__ wu_pk,
           const float* __restrict__ econst, const float* __restrict__ featv,
           const float* __restrict__ w_e, const int* __restrict__ seg,
           float* __restrict__ e1) {
  __shared__ __bf16 lds_a[2][MT * 16 * LDA];
  __shared__ float  e_part[2][MT * 16];
  __shared__ int    sseg[2][MT * 16];
  const int t = threadIdx.x;
  const int row0 = blockIdx.x * (2 * MT * 16);
  if (t < 128) { e_part[t >> 6][t & 63] = 0.0f; sseg[t >> 6][t & 63] = seg[row0 + t]; }

  // issue async staging for both macro-tiles (in-order completion)
  #pragma unroll
  for (int buf = 0; buf < 2; ++buf) {
    const unsigned lbase = (unsigned)(uintptr_t)&lds_a[buf][0];
    const char* gsrc = (const char*)(featb + (size_t)(row0 + buf * 64) * DIM);
    #pragma unroll
    for (int i = 0; i < 8; ++i) {               // 64 rows x 512B, 16B chunks
      int id = i * 256 + t;
      int row = id >> 5, c = (id & 31) * 16;
      async_copy16(lbase + row * (LDA * 2) + c, gsrc + row * 512 + c);
    }
  }

  wait_async8();            // tile 0 resident; tile 1 still streaming
  __syncthreads();
  gemm_u_compute(lds_a[0], sseg[0], e_part[0], wu_pk, econst, featv, w_e, t);
  wait_async0();            // tile 1 resident
  __syncthreads();
  if (t < 64) e1[row0 + t] = e_part[0][t];
  gemm_u_compute(lds_a[1], sseg[1], e_part[1], wu_pk, econst, featv, w_e, t);
  __syncthreads();
  if (t < 64) e1[row0 + 64 + t] = e_part[1][t];
}

// ---------------- Per-segment softmax over e1 -> alpha ---------------------
__global__ void k_seg_softmax(const float* __restrict__ e1, const int* __restrict__ seg_start,
                              float* __restrict__ alpha) {  // NSEG blocks x 256
  __shared__ float red[256];
  const int b = blockIdx.x, t = threadIdx.x;
  const int s = seg_start[b], e = seg_start[b + 1];
  float mx = -3.4e38f;
  for (int i = s + t; i < e; i += 256) mx = fmaxf(mx, e1[i]);
  red[t] = mx; __syncthreads();
  for (int o = 128; o > 0; o >>= 1) { if (t < o) red[t] = fmaxf(red[t], red[t + o]); __syncthreads(); }
  mx = red[0]; __syncthreads();
  float sum = 0.f;
  for (int i = s + t; i < e; i += 256) sum += __expf(e1[i] - mx);
  red[t] = sum; __syncthreads();
  for (int o = 128; o > 0; o >>= 1) { if (t < o) red[t] += red[t + o]; __syncthreads(); }
  float denom = red[0];
  for (int i = s + t; i < e; i += 256) alpha[i] = __expf(e1[i] - mx) / denom;
}

// ---------------- rst[b,:] = sum_{rows in b} (feat*scale+shift) * alpha ----
__global__ void k_seg_reduce(const float* __restrict__ feat, const float* __restrict__ scale,
                             const float* __restrict__ shift, const float* __restrict__ alpha,
                             const int* __restrict__ seg_start, float* __restrict__ rst) {
  const int b = blockIdx.x, c = threadIdx.x;  // NSEG blocks x 256
  const int s = seg_start[b], e = seg_start[b + 1];
  const float sc = scale[c], sh = shift[c];
  float acc = 0.f;
  for (int i = s; i < e; ++i)
    acc += (feat[(size_t)i * DIM + c] * sc + sh) * alpha[i];
  rst[(size_t)b * DIM + c] = acc;
}

// ---------------- out = rst @ w_out  (4096x256 @ 256x512) ------------------
__global__ void __launch_bounds__(256)
k_gemm_out(const float* __restrict__ rst, const __bf16* __restrict__ wo_pk,
           float* __restrict__ out) {   // 256 blocks x 256 thr
  __shared__ __bf16 lds_a[16 * LDA];
  const int rt = blockIdx.x, t = threadIdx.x;
  {
    const int rsub = t >> 6, c4 = (t & 63) * 4;
    #pragma unroll
    for (int i = 0; i < 4; ++i) {
      int r = i * 4 + rsub;
      const float4 f = *(const float4*)&rst[(size_t)(rt * 16 + r) * DIM + c4];
      v4bf bv; bv[0] = (__bf16)f.x; bv[1] = (__bf16)f.y; bv[2] = (__bf16)f.z; bv[3] = (__bf16)f.w;
      *(v4bf*)&lds_a[r * LDA + c4] = bv;
    }
  }
  __syncthreads();
  const int wave = t >> 5, lane = t & 31, hf = lane >> 4;
  #pragma unroll
  for (int i = 0; i < 4; ++i) {           // 32 n-tiles / 8 waves
    int nt = wave * 4 + i;
    v8f c = wmma_k256(lds_a, wo_pk + (size_t)nt * 8 * 512, lane);
    int n = nt * 16 + (lane & 15);
    #pragma unroll
    for (int j = 0; j < 8; ++j) {
      int rl = j + hf * 8;
      out[(size_t)(rt * 16 + rl) * OUTD + n] = c[j];
    }
  }
}

// ---------------------------------------------------------------------------
extern "C" void kernel_launch(void* const* d_in, const int* in_sizes, int n_in,
                              void* d_out, int out_size, void* d_ws, size_t ws_size,
                              hipStream_t stream) {
  const float* feat       = (const float*)d_in[0];
  const float* gamma      = (const float*)d_in[1];
  const float* beta       = (const float*)d_in[2];
  const float* w_u        = (const float*)d_in[3];
  const float* w_v        = (const float*)d_in[4];
  const float* b_v        = (const float*)d_in[5];
  const float* w_e        = (const float*)d_in[6];
  const float* w_out      = (const float*)d_in[7];
  const int*   seg_ids    = (const int*)d_in[8];
  const int*   last_nodes = (const int*)d_in[9];
  float* out = (float*)d_out;

  // workspace carve-up (bytes)
  char* ws = (char*)d_ws;
  size_t off = 0;
  float* sum    = (float*)(ws + off); off += DIM * 4;
  float* sumsq  = (float*)(ws + off); off += DIM * 4;
  float* scale  = (float*)(ws + off); off += DIM * 4;
  float* shift  = (float*)(ws + off); off += DIM * 4;
  float* econst = (float*)(ws + off); off += HID * 4;
  float* vconst = (float*)(ws + off); off += HID * 4;
  __bf16* wu_pk = (__bf16*)(ws + off); off += (size_t)DIM * HID * 2;
  __bf16* wv_pk = (__bf16*)(ws + off); off += (size_t)DIM * HID * 2;
  __bf16* wo_pk = (__bf16*)(ws + off); off += (size_t)DIM * OUTD * 2;
  float* featv  = (float*)(ws + off); off += (size_t)NSEG * HID * 4;
  float* e1     = (float*)(ws + off); off += (size_t)N_NODES * 4;
  float* alpha  = (float*)(ws + off); off += (size_t)N_NODES * 4;
  int* seg_start = (int*)(ws + off);  off += (NSEG + 1) * 4; off = (off + 511) & ~(size_t)511;
  float* rst    = (float*)(ws + off); off += (size_t)NSEG * DIM * 4;
  __bf16* featb = (__bf16*)(ws + off); off += (size_t)N_NODES * DIM * 2;  // bf16 feat copy

  // 1) BN stats (+ fused bf16 conversion of feat) -> scale/shift
  k_init_sums<<<1, 512, 0, stream>>>(sum);                       // zeros sum+sumsq
  k_bn_stats<<<N_NODES / 512, 256, 0, stream>>>(feat, sum, sumsq, featb);
  k_bn_finalize<<<1, 256, 0, stream>>>(sum, sumsq, gamma, beta, scale, shift);

  // 2) pack weights (BN scale folded into w_u/w_v rows); shift constants
  k_pack_b<<<(DIM * HID) / 256, 256, 0, stream>>>(w_u, scale, wu_pk, HID);
  k_pack_b<<<(DIM * HID) / 256, 256, 0, stream>>>(w_v, scale, wv_pk, HID);
  k_pack_b<<<(DIM * OUTD) / 256, 256, 0, stream>>>(w_out, (const float*)nullptr, wo_pk, OUTD);
  k_colconst<<<1, 256, 0, stream>>>(w_u, shift, econst, HID);
  k_colconst<<<1, 256, 0, stream>>>(w_v, shift, vconst, HID);

  // 3) segment boundaries
  k_seg_bounds<<<N_NODES / 256, 256, 0, stream>>>(seg_ids, seg_start);

  // 4) feat_v (async-staged gathered rows)
  k_gemm_v<<<NSEG / 16, 256, 0, stream>>>(featb, last_nodes, wv_pk, b_v, vconst, featv);

  // 5) big GEMM + gate -> e1 (double-buffered async staging)
  k_gemm_u_e<<<N_NODES / 128, 256, 0, stream>>>(featb, wu_pk, econst, featv, w_e,
                                                seg_ids, e1);

  // 6) segment softmax -> alpha
  k_seg_softmax<<<NSEG, 256, 0, stream>>>(e1, seg_start, alpha);

  // 7) segment weighted sum -> rst
  k_seg_reduce<<<NSEG, 256, 0, stream>>>(feat, scale, shift, alpha, seg_start, rst);

  // 8) out = rst @ w_out
  k_gemm_out<<<NSEG / 16, 256, 0, stream>>>(rst, wo_pk, out);
}
